// NCF_78752520339772
// MI455X (gfx1250) — matrix operand order repaired and checked by reading, not compiled, not run on hardware
//
#include <hip/hip_runtime.h>

typedef __attribute__((ext_vector_type(16))) _Float16 v16h;
typedef __attribute__((ext_vector_type(8)))  _Float16 v8h;
typedef __attribute__((ext_vector_type(8)))  float    v8f;

#define NCF_BATCH 16384
#define NCF_EMB   64

__device__ __forceinline__ float relu_f(float x) { return x > 0.f ? x : 0.f; }

// Fence same-wave LDS RAW (stores by some lanes -> loads by others).
__device__ __forceinline__ void lds_wave_sync() {
    asm volatile("s_wait_dscnt 0" ::: "memory");
}

// ---------------------------------------------------------------------------
// Weight pre-swizzle: f32 -> f16 in WMMA B-operand lane order.
//   B k order used here (contiguous, same as A): lane<16 holds column n0+lane, K = k0..k0+15
//                            lane>=16 holds column n0+lane-16, K = k0+16..k0+31
//   B1: [kc:4][nt:8][lane:32][j:16] halves   (W1: [128 out][128 in], h = x@W1^T)
//   B2: [kc:4][nt:4][lane:32][j:16] halves   (W2: [ 64 out][128 in])
// Each lane's per-WMMA B fetch becomes one contiguous 32B global load.
// ---------------------------------------------------------------------------
__global__ void ncf_prep_weights(const float* __restrict__ W1,
                                 const float* __restrict__ W2,
                                 _Float16* __restrict__ ws)
{
    int t = blockIdx.x * blockDim.x + threadIdx.x;
    if (t < 16384) {                      // B1
        int j    = t & 15;
        int lane = (t >> 4) & 31;
        int nt   = (t >> 9) & 7;
        int kc   = t >> 12;
        int n = nt * 16 + (lane & 15);
        int k = kc * 32 + ((lane >> 4) << 4) + j;
        ws[t] = (_Float16)W1[n * 128 + k];
    } else if (t < 24576) {               // B2
        int u = t - 16384;
        int j    = u & 15;
        int lane = (u >> 4) & 31;
        int nt   = (u >> 9) & 3;
        int kc   = u >> 11;
        int n = nt * 16 + (lane & 15);
        int k = kc * 32 + ((lane >> 4) << 4) + j;
        ws[t] = (_Float16)W2[n * 128 + k];
    }
}

// ---------------------------------------------------------------------------
// Fused NCF forward. 128 threads = 4 waves; each wave owns 16 samples.
// Lane mapping: col = lane&15 (sample / N-index), hi = lane>>4 (element half).
// ---------------------------------------------------------------------------
__global__ __launch_bounds__(128) void ncf_fused(
    const int* __restrict__ user, const int* __restrict__ item,
    const int* __restrict__ pre1, const int* __restrict__ pre2,
    const int* __restrict__ pre3, const int* __restrict__ pre4,
    const int* __restrict__ pre5,
    const float* __restrict__ Wu, const float* __restrict__ bu,
    const float* __restrict__ Wi, const float* __restrict__ bi,
    const float* __restrict__ b1, const float* __restrict__ b2,
    const float* __restrict__ W3, const float* __restrict__ b3,
    const _Float16* __restrict__ wsB,
    float* __restrict__ out)
{
    // Row stride 136 halves = 272B = 68 banks -> staggered, 16B aligned.
    __shared__ __align__(16) _Float16 xbuf[4][16 * 136];
    __shared__ __align__(16) _Float16 hbuf[4][16 * 136];

    const int lane = threadIdx.x & 31;
    const int wave = threadIdx.x >> 5;
    const int col  = lane & 15;
    const int hi   = lane >> 4;
    const int off  = hi * 32;                       // element chunk [off,off+32)
    const int b    = blockIdx.x * 64 + wave * 16 + col;

    // ---------------- gathers + attention (2 lanes per sample) -------------
    float4 biq[8];
    {
        const float4* Bi = (const float4*)(bi + off);
        #pragma unroll
        for (int q = 0; q < 8; ++q) biq[q] = Bi[q];
    }
    float ei[32];
    {
        const float4* Ei = (const float4*)(Wi + (size_t)item[b] * NCF_EMB + off);
        #pragma unroll
        for (int q = 0; q < 8; ++q) {
            float4 v = Ei[q];
            ei[4*q+0] = relu_f(v.x + biq[q].x);
            ei[4*q+1] = relu_f(v.y + biq[q].y);
            ei[4*q+2] = relu_f(v.z + biq[q].z);
            ei[4*q+3] = relu_f(v.w + biq[q].w);
        }
    }
    int pk[5];
    pk[0] = pre1[b]; pk[1] = pre2[b]; pk[2] = pre3[b]; pk[3] = pre4[b]; pk[4] = pre5[b];

    float w[5];
    #pragma unroll
    for (int k = 0; k < 5; ++k) {
        const float4* Ep = (const float4*)(Wi + (size_t)pk[k] * NCF_EMB + off);
        float part = 0.f;
        #pragma unroll
        for (int q = 0; q < 8; ++q) {
            float4 v = Ep[q];
            part += relu_f(v.x + biq[q].x) * ei[4*q+0];
            part += relu_f(v.y + biq[q].y) * ei[4*q+1];
            part += relu_f(v.z + biq[q].z) * ei[4*q+2];
            part += relu_f(v.w + biq[q].w) * ei[4*q+3];
        }
        w[k] = part + __shfl_xor(part, 16, 32);     // combine the two halves
    }
    float m = w[0];
    #pragma unroll
    for (int k = 1; k < 5; ++k) m = fmaxf(m, w[k]);
    float z[5], s = 0.f;
    #pragma unroll
    for (int k = 0; k < 5; ++k) { z[k] = __expf(w[k] - m); s += z[k]; }
    const float invs = 1.f / s;

    float pum[32];
    #pragma unroll
    for (int e = 0; e < 32; ++e) pum[e] = 0.f;
    #pragma unroll
    for (int k = 0; k < 5; ++k) {
        const float coef = z[k] * invs;
        const float4* Ep = (const float4*)(Wi + (size_t)pk[k] * NCF_EMB + off);
        #pragma unroll
        for (int q = 0; q < 8; ++q) {               // cache-hot re-read
            float4 v = Ep[q];
            pum[4*q+0] += coef * relu_f(v.x + biq[q].x);
            pum[4*q+1] += coef * relu_f(v.y + biq[q].y);
            pum[4*q+2] += coef * relu_f(v.z + biq[q].z);
            pum[4*q+3] += coef * relu_f(v.w + biq[q].w);
        }
    }

    // pu = relu(Wu[user]+bu) + 0.2*pum ; stage x = [pu | ei] as f16 in LDS.
    _Float16* xl = xbuf[wave];
    {
        const float4* Eu = (const float4*)(Wu + (size_t)user[b] * NCF_EMB + off);
        const float4* Bu = (const float4*)(bu + off);
        #pragma unroll
        for (int q = 0; q < 8; ++q) {
            float4 v = Eu[q], bq = Bu[q];
            xl[col*136 + off + 4*q + 0] = (_Float16)(relu_f(v.x + bq.x) + 0.2f * pum[4*q+0]);
            xl[col*136 + off + 4*q + 1] = (_Float16)(relu_f(v.y + bq.y) + 0.2f * pum[4*q+1]);
            xl[col*136 + off + 4*q + 2] = (_Float16)(relu_f(v.z + bq.z) + 0.2f * pum[4*q+2]);
            xl[col*136 + off + 4*q + 3] = (_Float16)(relu_f(v.w + bq.w) + 0.2f * pum[4*q+3]);
        }
    }
    #pragma unroll
    for (int e = 0; e < 32; ++e)
        xl[col*136 + 64 + off + e] = (_Float16)ei[e];
    lds_wave_sync();

    // ---------------- layer 1: h1 = relu(x @ W1^T + b1) via WMMA ------------
    _Float16* hl = hbuf[wave];
    #pragma unroll
    for (int nt = 0; nt < 8; ++nt) {
        v8f c = {};
        #pragma unroll
        for (int kc = 0; kc < 4; ++kc) {
            const int k0 = kc * 32;
            // A layout: lane<16 -> K k0..k0+15; lane>=16 -> K k0+16..k0+31
            v8h alo = *(const v8h*)(xl + col*136 + k0 + 16*hi);
            v8h ahi = *(const v8h*)(xl + col*136 + k0 + 16*hi + 8);
            v16h a  = __builtin_shufflevector(alo, ahi,
                        0,1,2,3,4,5,6,7,8,9,10,11,12,13,14,15);
            v16h bb = *(const v16h*)(wsB + (((kc*8 + nt)*32 + lane) << 4));
            c = __builtin_amdgcn_wmma_f32_16x16x32_f16(
                    false, a, false, bb, (short)0, c, false, false);
        }
        const float bias = b1[nt*16 + col];
        #pragma unroll
        for (int j = 0; j < 8; ++j)                 // C: M = j+8*hi, N = col
            hl[(j + 8*hi)*136 + nt*16 + col] = (_Float16)relu_f(c[j] + bias);
    }
    lds_wave_sync();

    // ---------------- layer 2 + head: sigmoid(relu(h1@W2^T+b2) @ W3^T + b3) -
    float acc[8];
    #pragma unroll
    for (int j = 0; j < 8; ++j) acc[j] = 0.f;
    #pragma unroll
    for (int nt = 0; nt < 4; ++nt) {
        v8f c = {};
        #pragma unroll
        for (int kc = 0; kc < 4; ++kc) {
            const int k0 = kc * 32;
            v8h alo = *(const v8h*)(hl + col*136 + k0 + 16*hi);
            v8h ahi = *(const v8h*)(hl + col*136 + k0 + 16*hi + 8);
            v16h a  = __builtin_shufflevector(alo, ahi,
                        0,1,2,3,4,5,6,7,8,9,10,11,12,13,14,15);
            v16h bb = *(const v16h*)(wsB + 16384 + (((kc*4 + nt)*32 + lane) << 4));
            c = __builtin_amdgcn_wmma_f32_16x16x32_f16(
                    false, a, false, bb, (short)0, c, false, false);
        }
        const float bias = b2[nt*16 + col];
        const float w3v  = W3[nt*16 + col];
        #pragma unroll
        for (int j = 0; j < 8; ++j)
            acc[j] += relu_f(c[j] + bias) * w3v;    // fold 64->1 dot into epilogue
    }
    const float bias3 = b3[0];
    #pragma unroll
    for (int j = 0; j < 8; ++j) {                   // reduce over N (16 lanes/half)
        float v = acc[j];
        v += __shfl_xor(v, 8, 32);
        v += __shfl_xor(v, 4, 32);
        v += __shfl_xor(v, 2, 32);
        v += __shfl_xor(v, 1, 32);
        if (col == j) {
            const float o = v + bias3;
            out[blockIdx.x*64 + wave*16 + j + 8*hi] = 1.f / (1.f + __expf(-o));
        }
    }
}

extern "C" void kernel_launch(void* const* d_in, const int* in_sizes, int n_in,
                              void* d_out, int out_size, void* d_ws, size_t ws_size,
                              hipStream_t stream)
{
    (void)in_sizes; (void)n_in; (void)out_size; (void)ws_size;
    const int*   user = (const int*)d_in[0];
    const int*   item = (const int*)d_in[1];
    const int*   pre1 = (const int*)d_in[2];
    const int*   pre2 = (const int*)d_in[3];
    const int*   pre3 = (const int*)d_in[4];
    const int*   pre4 = (const int*)d_in[5];
    const int*   pre5 = (const int*)d_in[6];
    const float* Wu   = (const float*)d_in[7];
    const float* bu   = (const float*)d_in[8];
    const float* Wi   = (const float*)d_in[9];
    const float* bi   = (const float*)d_in[10];
    const float* W1   = (const float*)d_in[11];
    const float* b1   = (const float*)d_in[12];
    const float* W2   = (const float*)d_in[13];
    const float* b2   = (const float*)d_in[14];
    const float* W3   = (const float*)d_in[15];
    const float* b3   = (const float*)d_in[16];
    _Float16* ws  = (_Float16*)d_ws;                // 24576 halves = 48 KB
    float*    out = (float*)d_out;

    ncf_prep_weights<<<96, 256, 0, stream>>>(W1, W2, ws);
    ncf_fused<<<NCF_BATCH / 64, 128, 0, stream>>>(
        user, item, pre1, pre2, pre3, pre4, pre5,
        Wu, bu, Wi, bi, b1, b2, W3, b3, ws, out);
}
